// TransformerDecoder_45114336477752
// MI455X (gfx1250) — matrix operand order repaired and checked by reading, not compiled, not run on hardware
//
#include <hip/hip_runtime.h>
#include <math.h>

typedef __attribute__((ext_vector_type(16))) __bf16 v16bf;
typedef __attribute__((ext_vector_type(8)))  __bf16 v8bf;
typedef __attribute__((ext_vector_type(8)))  float  v8f;
typedef __attribute__((ext_vector_type(4)))  float  v4f;

#define S_LEN   2048
#define D_MODEL 512
#define NHEAD   8
#define HEADD   64
#define NLAYER  12
#define VOCAB   256
#define FFDIM   2048

__device__ __forceinline__ __bf16 to_bf16(float f) { return (__bf16)f; }

// CDNA5 async global->LDS copy (raw 16B per lane), tracked by ASYNCcnt.
__device__ __forceinline__ void async_ld_b128(unsigned lds_off, unsigned long long gaddr) {
    asm volatile("global_load_async_to_lds_b128 %0, %1, off"
                 :: "v"(lds_off), "v"(gaddr) : "memory");
}
__device__ __forceinline__ void wait_async0() {
    asm volatile("s_wait_asynccnt 0" ::: "memory");
}

__device__ __forceinline__ v16bf concat8(v8bf lo, v8bf hi) {
    return __builtin_shufflevector(lo, hi, 0,1,2,3,4,5,6,7,8,9,10,11,12,13,14,15);
}

// ---------------------------------------------------------------------------
// x[s,d] = 2*WTE[idx[s],d] + pe[s,d]
// ---------------------------------------------------------------------------
__global__ __launch_bounds__(256)
void embed_kernel(const int* __restrict__ idx, const float* __restrict__ WTE,
                  const float* __restrict__ pe, float* __restrict__ x)
{
    int s = blockIdx.x;
    int tok = idx[s];
    for (int d = threadIdx.x; d < D_MODEL; d += 256) {
        x[(size_t)s * D_MODEL + d] =
            2.0f * WTE[(size_t)tok * D_MODEL + d] + pe[(size_t)s * D_MODEL + d];
    }
}

// ---------------------------------------------------------------------------
// Transpose + convert: WT[n][k] = bf16(W[k][n]).  Block 256 = 32x8, 32x32 tile.
// ---------------------------------------------------------------------------
__global__ __launch_bounds__(256)
void transpose_cvt_kernel(const float* __restrict__ W, __bf16* __restrict__ WT,
                          int K, int N)
{
    __shared__ float t[32][33];
    const int k0 = blockIdx.x * 32, n0 = blockIdx.y * 32;
    const int tx = threadIdx.x & 31, ty = threadIdx.x >> 5;
    #pragma unroll
    for (int i = 0; i < 4; ++i)
        t[ty + 8*i][tx] = W[(size_t)(k0 + ty + 8*i) * N + n0 + tx];
    __syncthreads();
    #pragma unroll
    for (int i = 0; i < 4; ++i)
        WT[(size_t)(n0 + ty + 8*i) * K + k0 + tx] = to_bf16(t[tx][ty + 8*i]);
}

// ---------------------------------------------------------------------------
// LayerNorm over D=512, one block (256 thr) per row, bf16 output.
// ---------------------------------------------------------------------------
__global__ __launch_bounds__(256)
void layernorm_kernel(const float* __restrict__ x, const float* __restrict__ g,
                      const float* __restrict__ b, __bf16* __restrict__ out)
{
    __shared__ float sred[256];
    const int s = blockIdx.x;
    const int t = threadIdx.x;
    const float* row = x + (size_t)s * D_MODEL;
    float v0 = row[t], v1 = row[t + 256];
    sred[t] = v0 + v1;
    __syncthreads();
    for (int o = 128; o > 0; o >>= 1) { if (t < o) sred[t] += sred[t + o]; __syncthreads(); }
    float mu = sred[0] * (1.0f / 512.0f);
    __syncthreads();
    float d0 = v0 - mu, d1 = v1 - mu;
    sred[t] = d0 * d0 + d1 * d1;
    __syncthreads();
    for (int o = 128; o > 0; o >>= 1) { if (t < o) sred[t] += sred[t + o]; __syncthreads(); }
    float rs = rsqrtf(sred[0] * (1.0f / 512.0f) + 1e-5f);
    out[(size_t)s * D_MODEL + t]       = to_bf16(d0 * rs * g[t] + b[t]);
    out[(size_t)s * D_MODEL + t + 256] = to_bf16(d1 * rs * g[t + 256] + b[t + 256]);
}

// ---------------------------------------------------------------------------
// C[M,N] = A[M,K](bf16) * WT[N,K](bf16) + bias (+ReLU) (+f32 residual).
// Block: 128 thr (4 waves), tile 64x128, wave tile 32x64.  BOTH tiles staged
// via global_load_async_to_lds_b128 double-buffered: zero ds-stores / zero
// converts in the K loop.  A tile stored in plain k order [0..31]
// so every A-frag is a single contiguous 32B ds_load.
// ---------------------------------------------------------------------------
template<bool RELU, bool RESID, bool OUTF, bool OUTB, bool OUTB_T>
__global__ __launch_bounds__(128)
void gemm_bf16_kernel(const __bf16* __restrict__ A, const __bf16* __restrict__ WT,
                      const float* __restrict__ bias, const float* __restrict__ resid,
                      float* __restrict__ outF, __bf16* __restrict__ outB,
                      int M, int N, int K)
{
    __shared__ __align__(64) __bf16 As[2][64][32];    //  8 KB (rows in k order)
    __shared__ __align__(64) __bf16 Bt[2][128][32];   // 16 KB, Bt[n][k]

    const int tid   = threadIdx.x;
    const int lane  = tid & 31;
    const int wid   = tid >> 5;
    const int half  = lane >> 4;
    const int l16   = lane & 15;
    const int bm    = blockIdx.x * 64;
    const int bn    = blockIdx.y * 128;
    const int waveM = (wid >> 1) * 32;
    const int waveN = (wid & 1) * 64;

    v8f acc[2][4];
    #pragma unroll
    for (int i = 0; i < 2; ++i)
        #pragma unroll
        for (int j = 0; j < 4; ++j) acc[i][j] = {};

    // A: two 16B runs per thread, destinations in k order so frag = 32B run.
    auto stageA = [&](int buf, int k0) {
        int r = tid >> 1, hb = tid & 1;                 // hb: k-half (0..15 / 16..31)
        const __bf16* src = A + (size_t)(bm + r) * K + k0 + hb * 16;
        unsigned lds = (unsigned)(unsigned long long)&As[buf][r][0];
        unsigned d0 = hb ? 32u : 0u;                    // k0..7->0, k16..23->32B
        unsigned d1 = hb ? 48u : 16u;                   // k8..15->16B, k24..31->48B
        async_ld_b128(lds + d0, (unsigned long long)src);
        async_ld_b128(lds + d1, (unsigned long long)(src + 8));
    };
    // B: one 64B row (4 x 16B) per thread, already transposed in global.
    auto stageB = [&](int buf, int k0) {
        int n = tid;
        const __bf16* src = WT + (size_t)(bn + n) * K + k0;
        unsigned lds = (unsigned)(unsigned long long)&Bt[buf][n][0];
        async_ld_b128(lds,      (unsigned long long)src);
        async_ld_b128(lds + 16, (unsigned long long)(src + 8));
        async_ld_b128(lds + 32, (unsigned long long)(src + 16));
        async_ld_b128(lds + 48, (unsigned long long)(src + 24));
    };

    const int nk = K >> 5;
    stageA(0, 0);
    stageB(0, 0);
    wait_async0();
    __syncthreads();

    for (int kt = 0; kt < nk; ++kt) {
        const int cur = kt & 1;
        if (kt + 1 < nk) {               // DMA next tile while computing
            stageA(cur ^ 1, (kt + 1) * 32);
            stageB(cur ^ 1, (kt + 1) * 32);
        }

        v16bf afrag[2], bfrag[4];
        #pragma unroll
        for (int tm = 0; tm < 2; ++tm)
            afrag[tm] = *(const v16bf*)&As[cur][waveM + tm * 16 + l16][16 * half];
        #pragma unroll
        for (int tn = 0; tn < 4; ++tn)
            bfrag[tn] = *(const v16bf*)&Bt[cur][waveN + tn * 16 + l16][16 * half];
        #pragma unroll
        for (int tm = 0; tm < 2; ++tm)
            #pragma unroll
            for (int tn = 0; tn < 4; ++tn)
                acc[tm][tn] = __builtin_amdgcn_wmma_f32_16x16x32_bf16(
                    false, afrag[tm], false, bfrag[tn], (short)0, acc[tm][tn], false, false);

        wait_async0();
        __syncthreads();
    }

    // --- epilogue ---
    #pragma unroll
    for (int tm = 0; tm < 2; ++tm)
        #pragma unroll
        for (int tn = 0; tn < 4; ++tn) {
            int n = bn + waveN + tn * 16 + l16;
            float bv = bias[n];
            int mbase = bm + waveM + tm * 16 + 8 * half;
            if constexpr (OUTB_T) {
                v8bf o;
                #pragma unroll
                for (int r = 0; r < 8; ++r) o[r] = to_bf16(acc[tm][tn][r] + bv);
                *(v8bf*)(outB + (size_t)n * M + mbase) = o;
            } else {
                #pragma unroll
                for (int r = 0; r < 8; ++r) {
                    int m = mbase + r;
                    float val = acc[tm][tn][r] + bv;
                    if constexpr (RELU)  val = fmaxf(val, 0.0f);
                    if constexpr (RESID) val += resid[(size_t)m * N + n];
                    if constexpr (OUTF)  outF[(size_t)m * N + n] = val;
                    if constexpr (OUTB)  outB[(size_t)m * N + n] = to_bf16(val);
                }
            }
        }
}

// ---------------------------------------------------------------------------
// Flash-style attention, NO causal mask (matches reference).
// Block = (16-row q-tile, head), 128 threads (4 waves). Key chunks of 512:
// scores f32 in LDS -> online softmax -> probs written as bf16 (plain k order)
// -> P@V with two independent WMMA accumulator chains. V pre-transposed.
// ---------------------------------------------------------------------------
__global__ __launch_bounds__(128)
void attention_kernel(const __bf16* __restrict__ Q, const __bf16* __restrict__ Kv,
                      const __bf16* __restrict__ VT, __bf16* __restrict__ O)
{
    __shared__ __align__(64) float   sc[16][512];   // 32 KB raw scores
    __shared__ __align__(64) __bf16  pb[16][512];   // 16 KB probs (plain k order)
    __shared__ float red[16][8];
    __shared__ float rowMax[16], rowSum[16], rowScale[16];

    const int tid   = threadIdx.x;
    const int lane  = tid & 31;
    const int wid   = tid >> 5;
    const int half  = lane >> 4;
    const int l16   = lane & 15;
    const int qbase = blockIdx.x * 16;
    const int hof   = blockIdx.y * HEADD;
    const float scale = 0.125f; // 1/sqrt(64)

    if (tid < 16) { rowMax[tid] = -1e30f; rowSum[tid] = 0.0f; }

    // Q A-frags (16x64 -> two K=32 halves), built once from global
    v16bf qa[2];
    {
        const __bf16* qrow = Q + (size_t)(qbase + l16) * D_MODEL + hof;
        #pragma unroll
        for (int kh = 0; kh < 2; ++kh) {
            v8bf lo = *(const v8bf*)(qrow + 32 * kh + 8 * half);
            v8bf hi = *(const v8bf*)(qrow + 32 * kh + 16 + 8 * half);
            qa[kh] = concat8(lo, hi);
        }
    }

    v8f oacc0 = {}, oacc1 = {};
    __syncthreads();

    for (int c = 0; c < S_LEN / 512; ++c) {
        const int kchunk = c * 512;

        // Phase A: scores for this chunk (8 key-tiles per wave)
        for (int i = 0; i < 8; ++i) {
            int kt   = i * 4 + wid;
            int keyb = kchunk + kt * 16;
            v8f s = {};
            #pragma unroll
            for (int kh = 0; kh < 2; ++kh) {
                const __bf16* krow = Kv + (size_t)(keyb + l16) * D_MODEL + hof + 32 * kh;
                v16bf b = concat8(*(const v8bf*)(krow + 8 * half), *(const v8bf*)(krow + 16 + 8 * half));
                s = __builtin_amdgcn_wmma_f32_16x16x32_bf16(
                        false, qa[kh], false, b, (short)0, s, false, false);
            }
            #pragma unroll
            for (int r = 0; r < 8; ++r)
                sc[r + 8 * half][kt * 16 + l16] = s[r] * scale;
        }
        __syncthreads();

        // Phase B1: chunk row-max (8 threads per row, v4f scans)
        {
            int row = tid >> 3, sub = tid & 7;
            const v4f* p4 = (const v4f*)(sc[row] + sub * 64);
            float mx = -1e30f;
            #pragma unroll
            for (int i = 0; i < 16; ++i) {
                v4f w = p4[i];
                mx = fmaxf(mx, fmaxf(fmaxf(w[0], w[1]), fmaxf(w[2], w[3])));
            }
            red[row][sub] = mx;
        }
        __syncthreads();
        if (tid < 16) {
            float mx = red[tid][0];
            for (int i = 1; i < 8; ++i) mx = fmaxf(mx, red[tid][i]);
            float newM = fmaxf(rowMax[tid], mx);
            rowScale[tid] = __expf(rowMax[tid] - newM);
            rowMax[tid] = newM;
        }
        __syncthreads();

        // Phase B2: exp -> bf16 probs (plain k order), + sums
        {
            int row = tid >> 3, sub = tid & 7;
            float m = rowMax[row], ssum = 0.0f;
            const v8f* p8 = (const v8f*)(sc[row] + sub * 64);
            #pragma unroll
            for (int i = 0; i < 8; ++i) {
                v8f w = p8[i];
                v8bf e;
                #pragma unroll
                for (int j = 0; j < 8; ++j) {
                    float ev = __expf(w[j] - m);
                    e[j] = to_bf16(ev);
                    ssum += ev;
                }
                int base = sub * 64 + 8 * i;
                int off  = base & 31;                    // 0,8,16,24
                int poff = off;
                *(v8bf*)&pb[row][(base & ~31) | poff] = e;
            }
            red[row][sub] = ssum;
        }
        __syncthreads();
        if (tid < 16) {
            float ssum = 0.0f;
            for (int i = 0; i < 8; ++i) ssum += red[tid][i];
            rowSum[tid] = rowSum[tid] * rowScale[tid] + ssum;
        }
        __syncthreads();

        // Phase C: rescale both chains, then O += P(chunk) @ V(chunk)
        #pragma unroll
        for (int r = 0; r < 8; ++r) {
            float f = rowScale[r + 8 * half];
            oacc0[r] *= f;
            oacc1[r] *= f;
        }

        const __bf16* vbase = VT + (size_t)(hof + wid * 16 + l16) * S_LEN + kchunk;
        for (int s0 = 0; s0 < 16; s0 += 2) {
            v16bf a0 = *(const v16bf*)&pb[l16][32 * s0 + 16 * half];
            v16bf b0 = *(const v16bf*)(vbase + 32 * s0 + 16 * half);
            oacc0 = __builtin_amdgcn_wmma_f32_16x16x32_bf16(
                        false, a0, false, b0, (short)0, oacc0, false, false);

            v16bf a1 = *(const v16bf*)&pb[l16][32 * (s0 + 1) + 16 * half];
            v16bf b1 = *(const v16bf*)(vbase + 32 * (s0 + 1) + 16 * half);
            oacc1 = __builtin_amdgcn_wmma_f32_16x16x32_bf16(
                        false, a1, false, b1, (short)0, oacc1, false, false);
        }
        __syncthreads();
    }

    #pragma unroll
    for (int r = 0; r < 8; ++r) {
        int m = r + 8 * half;
        O[(size_t)(qbase + m) * D_MODEL + hof + wid * 16 + l16] =
            to_bf16((oacc0[r] + oacc1[r]) / rowSum[m]);
    }
}

// ---------------------------------------------------------------------------
extern "C" void kernel_launch(void* const* d_in, const int* in_sizes, int n_in,
                              void* d_out, int out_size, void* d_ws, size_t ws_size,
                              hipStream_t stream)
{
    const int*   idx = (const int*)  d_in[0];
    const float* WTE = (const float*)d_in[1];
    const float* pe  = (const float*)d_in[2];
    const float* Wq  = (const float*)d_in[3];
    const float* bq  = (const float*)d_in[4];
    const float* Wk  = (const float*)d_in[5];
    const float* bk  = (const float*)d_in[6];
    const float* Wv  = (const float*)d_in[7];
    const float* bv  = (const float*)d_in[8];
    const float* Wo  = (const float*)d_in[9];
    const float* bo  = (const float*)d_in[10];
    const float* g1  = (const float*)d_in[11];
    const float* be1 = (const float*)d_in[12];
    const float* W1  = (const float*)d_in[13];
    const float* fb1 = (const float*)d_in[14];
    const float* W2  = (const float*)d_in[15];
    const float* fb2 = (const float*)d_in[16];
    const float* g2  = (const float*)d_in[17];
    const float* be2 = (const float*)d_in[18];
    const float* gF  = (const float*)d_in[19];
    const float* bF  = (const float*)d_in[20];
    const float* Wh  = (const float*)d_in[21];
    const float* bh  = (const float*)d_in[22];

    char* ws = (char*)d_ws;
    float*  x      = (float*) (ws);                               //  4 MB
    __bf16* h_bf   = (__bf16*)(ws +  4u * 1024 * 1024);           //  2 MB
    __bf16* q_bf   = (__bf16*)(ws +  6u * 1024 * 1024);           //  2 MB
    __bf16* k_bf   = (__bf16*)(ws +  8u * 1024 * 1024);           //  2 MB
    __bf16* v_bfT  = (__bf16*)(ws + 10u * 1024 * 1024);           //  2 MB (VT[d][s])
    __bf16* o_bf   = (__bf16*)(ws + 12u * 1024 * 1024);           //  2 MB
    __bf16* mid_bf = (__bf16*)(ws + 14u * 1024 * 1024);           //  8 MB
    __bf16* wT     = (__bf16*)(ws + 22u * 1024 * 1024);           // 72.25 MB weights

    // bf16 transposed weight region: per layer 3,145,728 elements (6 MB)
    const size_t DD = (size_t)D_MODEL * D_MODEL;   // 262144
    const size_t DF = (size_t)D_MODEL * FFDIM;     // 1048576
    const size_t LSTRIDE = 4 * DD + 2 * DF;        // 3145728
    __bf16* whT = wT + NLAYER * LSTRIDE;           // [VOCAB][D_MODEL]

    // --- one-time (per launch) weight convert+transpose: WT[n][k] ---
    for (int l = 0; l < NLAYER; ++l) {
        const size_t wOff = (size_t)l * DD;
        __bf16* base = wT + (size_t)l * LSTRIDE;
        transpose_cvt_kernel<<<dim3(D_MODEL/32, D_MODEL/32), 256, 0, stream>>>(Wq + wOff, base + 0*DD, D_MODEL, D_MODEL);
        transpose_cvt_kernel<<<dim3(D_MODEL/32, D_MODEL/32), 256, 0, stream>>>(Wk + wOff, base + 1*DD, D_MODEL, D_MODEL);
        transpose_cvt_kernel<<<dim3(D_MODEL/32, D_MODEL/32), 256, 0, stream>>>(Wv + wOff, base + 2*DD, D_MODEL, D_MODEL);
        transpose_cvt_kernel<<<dim3(D_MODEL/32, D_MODEL/32), 256, 0, stream>>>(Wo + wOff, base + 3*DD, D_MODEL, D_MODEL);
        transpose_cvt_kernel<<<dim3(D_MODEL/32, FFDIM/32),   256, 0, stream>>>(W1 + (size_t)l*DF, base + 4*DD, D_MODEL, FFDIM);
        transpose_cvt_kernel<<<dim3(FFDIM/32, D_MODEL/32),   256, 0, stream>>>(W2 + (size_t)l*DF, base + 4*DD + DF, FFDIM, D_MODEL);
    }
    transpose_cvt_kernel<<<dim3(D_MODEL/32, VOCAB/32), 256, 0, stream>>>(Wh, whT, D_MODEL, VOCAB);

    const dim3 gemmBlk(128);
    const dim3 lnGrid(S_LEN), lnBlk(256);

    embed_kernel<<<S_LEN, 256, 0, stream>>>(idx, WTE, pe, x);

    for (int l = 0; l < NLAYER; ++l) {
        __bf16* base = wT + (size_t)l * LSTRIDE;

        layernorm_kernel<<<lnGrid, lnBlk, 0, stream>>>(x, g1 + l * D_MODEL, be1 + l * D_MODEL, h_bf);

        gemm_bf16_kernel<false,false,false,true,false><<<dim3(S_LEN/64, D_MODEL/128), gemmBlk, 0, stream>>>(
            h_bf, base + 0*DD, bq + l * D_MODEL, nullptr, nullptr, q_bf, S_LEN, D_MODEL, D_MODEL);
        gemm_bf16_kernel<false,false,false,true,false><<<dim3(S_LEN/64, D_MODEL/128), gemmBlk, 0, stream>>>(
            h_bf, base + 1*DD, bk + l * D_MODEL, nullptr, nullptr, k_bf, S_LEN, D_MODEL, D_MODEL);
        // V projection written TRANSPOSED: VT[d][s]
        gemm_bf16_kernel<false,false,false,false,true><<<dim3(S_LEN/64, D_MODEL/128), gemmBlk, 0, stream>>>(
            h_bf, base + 2*DD, bv + l * D_MODEL, nullptr, nullptr, v_bfT, S_LEN, D_MODEL, D_MODEL);

        attention_kernel<<<dim3(S_LEN/16, NHEAD), dim3(128), 0, stream>>>(q_bf, k_bf, v_bfT, o_bf);

        // x = x + O @ Wo + bo   (in-place residual)
        gemm_bf16_kernel<false,true,true,false,false><<<dim3(S_LEN/64, D_MODEL/128), gemmBlk, 0, stream>>>(
            o_bf, base + 3*DD, bo + l * D_MODEL, x, x, nullptr, S_LEN, D_MODEL, D_MODEL);

        layernorm_kernel<<<lnGrid, lnBlk, 0, stream>>>(x, g2 + l * D_MODEL, be2 + l * D_MODEL, h_bf);

        // mid = relu(h @ W1 + fb1) -> bf16
        gemm_bf16_kernel<true,false,false,true,false><<<dim3(S_LEN/64, FFDIM/128), gemmBlk, 0, stream>>>(
            h_bf, base + 4*DD, fb1 + l * FFDIM, nullptr, nullptr, mid_bf, S_LEN, FFDIM, D_MODEL);

        // x = x + mid @ W2 + fb2
        gemm_bf16_kernel<false,true,true,false,false><<<dim3(S_LEN/64, D_MODEL/128), gemmBlk, 0, stream>>>(
            mid_bf, base + 4*DD + DF, fb2 + l * D_MODEL, x, x, nullptr, S_LEN, D_MODEL, FFDIM);
    }

    layernorm_kernel<<<lnGrid, lnBlk, 0, stream>>>(x, gF, bF, h_bf);

    gemm_bf16_kernel<false,false,true,false,false><<<dim3(S_LEN/64, VOCAB/128), gemmBlk, 0, stream>>>(
        h_bf, whT, bh, nullptr, (float*)d_out, nullptr, S_LEN, VOCAB, D_MODEL);
}
